// ResidualPointMLP2d_13159779795312
// MI455X (gfx1250) — matrix operand order repaired and checked by reading, not compiled, not run on hardware
//
#include <hip/hip_runtime.h>
#include <hip/hip_bf16.h>
#include <math.h>

// ---------------------------------------------------------------------------
// ResidualPointMLP2d for MI455X (gfx1250).
// Two fused per-pixel GEMMs (65536 pixels): W1*X^T -> LN(128) -> GELU ->
// W2*h -> LN(256) -> +x -> GELU.  Memory roofline: 134 MB compulsory HBM @
// 23.3 TB/s ~= 5.8us; bf16 WMMA compute ~= 1.7us.  Weights are pre-converted
// and pre-swizzled into d_ws (zero conversion ALU in the hot loop); GELU uses
// an A&S 7.1.26 erf (abs err <= 1.5e-7, ~13 VALU + rcp/exp TRANS ops) so the
// epilogue ALU stays under the memory roofline.
// ---------------------------------------------------------------------------

typedef __attribute__((ext_vector_type(16))) __bf16 v16bf;
typedef __attribute__((ext_vector_type(8)))  float  v8f;

union Frag {
  v16bf v;
  unsigned int ui[8];
  uint4 q[2];
};

// Pack two f32 -> bf16 pair: +0x8000 nearest-bias, then one v_perm_b32.
__device__ __forceinline__ unsigned int pk2bf(float lo, float hi) {
  unsigned int a = __float_as_uint(lo) + 0x8000u;
  unsigned int b = __float_as_uint(hi) + 0x8000u;
  return __builtin_amdgcn_perm(b, a, 0x07060302u);  // [lo.hi16, hi.hi16]
}

// Exact-erf GELU via Abramowitz-Stegun 7.1.26: |err(erf)| <= 1.5e-7,
// using hardware v_rcp_f32 / v_exp_f32 (TRANS ops co-execute with VALU).
__device__ __forceinline__ float gelu_erf(float x) {
  const float z = fabsf(x) * 0.70710678118654752440f;   // |x|/sqrt(2)
  const float k = __builtin_amdgcn_rcpf(fmaf(0.3275911f, z, 1.0f));
  float p = fmaf(1.061405429f, k, -1.453152027f);
  p = fmaf(p, k, 1.421413741f);
  p = fmaf(p, k, -0.284496736f);
  p = fmaf(p, k, 0.254829592f);
  p = p * k;
  const float e  = __expf(-z * z);                      // v_exp_f32
  const float er = fmaf(-p, e, 1.0f);                   // erf(|x|/sqrt2)
  const float es = copysignf(er, x);
  return 0.5f * x * (1.0f + es);
}

__device__ __forceinline__ v8f zero8() {
  v8f z;
#pragma unroll
  for (int i = 0; i < 8; ++i) z[i] = 0.f;
  return z;
}
__device__ __forceinline__ void ld4(const float* __restrict__ p, float o[4]) {
  const float4 t = *reinterpret_cast<const float4*>(p);
  o[0] = t.x; o[1] = t.y; o[2] = t.z; o[3] = t.w;
}
// A-fragment K index for 16-bit WMMA A layout (16x32), VGPR pair j, lane group lg
__device__ __forceinline__ int a_k(int j, int lg) {
  return 2 * j + (j >= 4 ? 8 : 0) + lg * 8;
}

#define C_IN 256
#define C_H  128
#define HW   4096
#define WAVES 8
#define HPAD 132
// ws layout (uints): W1 frags [f=n*8+kk][256 uints], then W2 frags [f=n*4+kk][256]
#define WS_W2_OFF 16384   // 64 frags * 256 uints

// ---- prep: fp32 weights -> bf16, pre-swizzled into per-lane contiguous frags
__global__ __launch_bounds__(256)
void prep_weights(const float* __restrict__ W1, const float* __restrict__ W2,
                  unsigned int* __restrict__ ws) {
  const int tid  = blockIdx.x * 256 + threadIdx.x;   // 4096 threads
  const int lane = tid & 31;
  const int lr = lane & 15, lg = lane >> 4;
  if (tid < 2048) {                                  // W1: 64 frags
    const int f = tid >> 5, n = f >> 3, kk = f & 7;
    unsigned int* dst = ws + f * 256;
    const float* src = W1 + (n * 16 + lr) * C_IN + kk * 32;
#pragma unroll
    for (int j = 0; j < 8; ++j) {
      const int k = a_k(j, lg);
      const unsigned int p = pk2bf(src[k], src[k + 1]);
      if (j < 4) dst[lane * 4 + j]           = p;    // q[0] half: lane-contig
      else       dst[128 + lane * 4 + j - 4] = p;    // q[1] half
    }
  } else {                                           // W2: 64 frags
    const int t = tid - 2048;
    const int f = t >> 5, n = f >> 2, kk = f & 3;
    unsigned int* dst = ws + WS_W2_OFF + f * 256;
    const float* src = W2 + (n * 16 + lr) * C_H + kk * 32;
#pragma unroll
    for (int j = 0; j < 8; ++j) {
      const int k = a_k(j, lg);
      const unsigned int p = pk2bf(src[k], src[k + 1]);
      if (j < 4) dst[lane * 4 + j]           = p;
      else       dst[128 + lane * 4 + j - 4] = p;
    }
  }
}

__global__ __launch_bounds__(256)
void fused_residual_mlp(const float* __restrict__ x,
                        const float* __restrict__ b1,
                        const float* __restrict__ g1, const float* __restrict__ be1,
                        const float* __restrict__ b2,
                        const float* __restrict__ g2, const float* __restrict__ be2,
                        const unsigned int* __restrict__ wsw,
                        float* __restrict__ out) {
  __shared__ __align__(16) unsigned short hs[WAVES][16 * HPAD];

  const int lane = threadIdx.x & 31;
  const int wave = threadIdx.x >> 5;
  const int lr   = lane & 15;   // pixel within wave tile (B/D column)
  const int lg   = lane >> 4;

  const int pbase = blockIdx.x * (WAVES * 16) + wave * 16;
  const int bimg  = pbase >> 12;
  const int hw0   = pbase & 4095;
  const float* xw = x   + (size_t)bimg * (C_IN * HW) + hw0;
  float*       ow = out + (size_t)bimg * (C_IN * HW) + hw0;

  // ====================== GEMM1: h[128 x 16] = W1 * X^T =====================
  v8f acc1[8];
#pragma unroll
  for (int n = 0; n < 8; ++n) acc1[n] = zero8();

#pragma unroll 1
  for (int kk = 0; kk < 8; ++kk) {
    Frag bx;                                   // X^T fragment (convert once)
#pragma unroll
    for (int j = 0; j < 8; ++j) {
      const int k = kk * 32 + a_k(j, lg);
      bx.ui[j] = pk2bf(xw[(size_t)k * HW + lr], xw[(size_t)(k + 1) * HW + lr]);
    }
#pragma unroll
    for (int n = 0; n < 8; ++n) {
      Frag aw;                                 // pre-swizzled bf16 weights
      const uint4* pa = reinterpret_cast<const uint4*>(wsw + (size_t)(n * 8 + kk) * 256);
      aw.q[0] = pa[lane];
      aw.q[1] = pa[lane + 32];
      acc1[n] = __builtin_amdgcn_wmma_f32_16x16x32_bf16(
          false, aw.v, false, bx.v, (short)0, acc1[n], false, false);
    }
  }

  // ============== Epilogue1: +b1, LN over 128 ch, GELU, -> LDS =============
  float s = 0.f, q = 0.f;
#pragma unroll
  for (int n = 0; n < 8; ++n) {
    float bv[8];
    ld4(b1 + n * 16 + lg * 8, bv);
    ld4(b1 + n * 16 + lg * 8 + 4, bv + 4);
#pragma unroll
    for (int i = 0; i < 8; ++i) {
      const float v = acc1[n][i] + bv[i];
      acc1[n][i] = v;
      s += v; q += v * v;
    }
  }
  s += __shfl_xor(s, 16, 32);
  q += __shfl_xor(q, 16, 32);
  float mean = s * (1.f / 128.f);
  float rstd = rsqrtf(q * (1.f / 128.f) - mean * mean + 1e-5f);
#pragma unroll
  for (int n = 0; n < 8; ++n) {
    float gv[8], bev[8];
    ld4(g1 + n * 16 + lg * 8, gv);  ld4(g1 + n * 16 + lg * 8 + 4, gv + 4);
    ld4(be1 + n * 16 + lg * 8, bev); ld4(be1 + n * 16 + lg * 8 + 4, bev + 4);
#pragma unroll
    for (int i2 = 0; i2 < 4; ++i2) {
      const float v0 = gelu_erf((acc1[n][2 * i2]     - mean) * rstd * gv[2 * i2]     + bev[2 * i2]);
      const float v1 = gelu_erf((acc1[n][2 * i2 + 1] - mean) * rstd * gv[2 * i2 + 1] + bev[2 * i2 + 1]);
      const int oc = n * 16 + lg * 8 + 2 * i2;          // even -> 4B aligned
      *reinterpret_cast<unsigned int*>(&hs[wave][lr * HPAD + oc]) = pk2bf(v0, v1);
    }
  }
  __syncthreads();

  // ====================== GEMM2: y[256 x 16] = W2 * h ======================
  v8f acc2[16];
#pragma unroll
  for (int n = 0; n < 16; ++n) acc2[n] = zero8();

  const unsigned short* hrow = &hs[wave][0];
#pragma unroll 1
  for (int kk = 0; kk < 4; ++kk) {
    Frag bh;
#pragma unroll
    for (int j = 0; j < 8; ++j) {
      const int k = kk * 32 + a_k(j, lg);
      bh.ui[j] = *reinterpret_cast<const unsigned int*>(&hrow[lr * HPAD + k]);
    }
#pragma unroll
    for (int n = 0; n < 16; ++n) {
      Frag aw;
      const uint4* pa =
          reinterpret_cast<const uint4*>(wsw + WS_W2_OFF + (size_t)(n * 4 + kk) * 256);
      aw.q[0] = pa[lane];
      aw.q[1] = pa[lane + 32];
      acc2[n] = __builtin_amdgcn_wmma_f32_16x16x32_bf16(
          false, aw.v, false, bh.v, (short)0, acc2[n], false, false);
    }
  }

  // ========= Epilogue2: +b2, LN over 256 ch, +residual, GELU, store ========
  s = 0.f; q = 0.f;
#pragma unroll
  for (int n = 0; n < 16; ++n) {
    float bv[8];
    ld4(b2 + n * 16 + lg * 8, bv);
    ld4(b2 + n * 16 + lg * 8 + 4, bv + 4);
#pragma unroll
    for (int i = 0; i < 8; ++i) {
      const float v = acc2[n][i] + bv[i];
      acc2[n][i] = v;
      s += v; q += v * v;
    }
  }
  s += __shfl_xor(s, 16, 32);
  q += __shfl_xor(q, 16, 32);
  mean = s * (1.f / 256.f);
  rstd = rsqrtf(q * (1.f / 256.f) - mean * mean + 1e-5f);
#pragma unroll
  for (int n = 0; n < 16; ++n) {
    float gv[8], bev[8];
    ld4(g2 + n * 16 + lg * 8, gv);  ld4(g2 + n * 16 + lg * 8 + 4, gv + 4);
    ld4(be2 + n * 16 + lg * 8, bev); ld4(be2 + n * 16 + lg * 8 + 4, bev + 4);
#pragma unroll
    for (int i = 0; i < 8; ++i) {
      const int oc = n * 16 + lg * 8 + i;
      const float v = (acc2[n][i] - mean) * rstd * gv[i] + bev[i];
      // residual re-read hits L2 (x fits in 192MB); lanes 0-15 contiguous
      const float r = v + xw[(size_t)oc * HW + lr];
      ow[(size_t)oc * HW + lr] = gelu_erf(r);
    }
  }
}

extern "C" void kernel_launch(void* const* d_in, const int* in_sizes, int n_in,
                              void* d_out, int out_size, void* d_ws, size_t ws_size,
                              hipStream_t stream) {
  (void)in_sizes; (void)n_in; (void)out_size; (void)ws_size;  // needs >=128KB ws
  const float* x   = (const float*)d_in[0];
  const float* W1  = (const float*)d_in[1];
  const float* b1  = (const float*)d_in[2];
  const float* g1  = (const float*)d_in[3];
  const float* be1 = (const float*)d_in[4];
  const float* W2  = (const float*)d_in[5];
  const float* b2  = (const float*)d_in[6];
  const float* g2  = (const float*)d_in[7];
  const float* be2 = (const float*)d_in[8];
  unsigned int* ws = (unsigned int*)d_ws;

  prep_weights<<<16, 256, 0, stream>>>(W1, W2, ws);
  // 65536 pixels; 128 per 256-thread block (8 wave32 x 16 pixels)
  fused_residual_mlp<<<512, 256, 0, stream>>>(x, b1, g1, be1, b2, g2, be2,
                                              ws, (float*)d_out);
}
